// SentenceClassifier_45878840655907
// MI455X (gfx1250) — compile-verified, matrix operand order checked by reading, not run on hardware
//
#include <hip/hip_runtime.h>
#include <hip/hip_bf16.h>

// ---------------------------------------------------------------------------
// Sentence classifier on gfx1250 (MI455X): bf16 WMMA for all GEMMs,
// fused LSTM step kernels with LDS-staged weights (320KB WGP LDS),
// algebraically-collapsed attention.
// ---------------------------------------------------------------------------

typedef __attribute__((ext_vector_type(16))) __bf16 v16bf;
typedef __attribute__((ext_vector_type(8)))  float  v8f;

#define NN    256   // batch of sentences
#define LLEN  128   // tokens per sentence
#define DD    300   // embedding dim
#define DP    320   // padded embedding dim (multiple of 32 for WMMA K)
#define EE    512   // LSTM hidden
#define G4    2048  // 4*EE (gates)
#define AHID  256   // attention hidden
#define HOPS  4
#define FF    4096  // 2E * HOP sentence feature
#define NLAB  7

// padded LDS row strides (elements) to avoid 256B-period bank conflicts
#define HH_STR (EE + 16)   // 528
#define IH_STR (DP + 16)   // 336
#define LDS_BYTES ((4 * 16 * HH_STR + 4 * 16 * IH_STR) * 2)  // 110592

__device__ __forceinline__ unsigned short f2bf(float f) {
  unsigned int u = __float_as_uint(f);
  u += 0x7fffu + ((u >> 16) & 1u);          // round-to-nearest-even
  return (unsigned short)(u >> 16);
}
__device__ __forceinline__ float bf2f(unsigned short h) {
  return __uint_as_float(((unsigned int)h) << 16);
}
__device__ __forceinline__ float sigm(float x) { return 1.0f / (1.0f + __expf(-x)); }

// Load one lane's A/B fragment slice for v_wmma_f32_16x16x32_bf16.
// Lane holds k = [bk, bk+8) in elements 0..7 and [bk+16, bk+24) in 8..15,
// where bk = kchunk*32 + (lane>=16 ? 8 : 0). Two 16B contiguous loads.
__device__ __forceinline__ v16bf load_frag(const unsigned short* base) {
  v16bf a;
  uint4* p = reinterpret_cast<uint4*>(&a);
  p[0] = *reinterpret_cast<const uint4*>(base);
  p[1] = *reinterpret_cast<const uint4*>(base + 16);
  return a;
}

// ---------------------------------------------------------------------------
// small utility kernels
// ---------------------------------------------------------------------------
__global__ void fill_zero_u32(unsigned int* __restrict__ p, long n) {
  long id = (long)blockIdx.x * blockDim.x + threadIdx.x;
  if (id < n) p[id] = 0u;
}

__global__ void cvt_bf16(const float* __restrict__ src, unsigned short* __restrict__ dst, long n) {
  long id = (long)blockIdx.x * blockDim.x + threadIdx.x;
  if (id < n) dst[id] = f2bf(src[id]);
}

// [rows, DD] f32 -> [rows, DP] bf16 with zero pad
__global__ void cvt_pad_bf16(const float* __restrict__ src, unsigned short* __restrict__ dst, int rows) {
  long id = (long)blockIdx.x * blockDim.x + threadIdx.x;
  if (id >= (long)rows * DP) return;
  int k = (int)(id % DP);
  long r = id / DP;
  float v = (k < DD) ? src[r * DD + k] : 0.0f;
  dst[id] = f2bf(v);
}

__global__ void add_vec(const float* __restrict__ a, const float* __restrict__ b,
                        float* __restrict__ dst, int n) {
  int id = blockIdx.x * blockDim.x + threadIdx.x;
  if (id < n) dst[id] = a[id] + b[id];
}

// tokens -> padded bf16 embeddings [NN, LLEN, DP]
__global__ void embed_pad(const int* __restrict__ tokens, const float* __restrict__ table,
                          unsigned short* __restrict__ emb) {
  long id = (long)blockIdx.x * blockDim.x + threadIdx.x;
  if (id >= (long)NN * LLEN * DP) return;
  int k = (int)(id % DP);
  long nl = id / DP;
  float v = 0.0f;
  if (k < DD) v = table[(long)tokens[nl] * DD + k];
  emb[id] = f2bf(v);
}

// ---------------------------------------------------------------------------
// Fused token-LSTM step: for time t, g = x_t@Wih^T + h@Whh^T + bias, then
// gate nonlinearities and c/h update.
//
// Block = 256 threads = 8 waves. blockIdx.x = jt (gate-local 16-col group),
// blockIdx.y selects a group of 8 M-tiles; wave w handles mt = 8*y + w.
// The block's weight slice (4 gates x 16 rows of Whh and Wih, ~108KB) is
// cooperatively staged into LDS ONCE and reused by all 8 M-tile waves,
// cutting the dominant L2 weight traffic 8x versus direct global loads.
// Grid: (32, 2) x 256 threads = 512 waves = 16 m-tiles x 32 j-tiles.
// ---------------------------------------------------------------------------
__global__ void tok_lstm_step(const unsigned short* __restrict__ emb,   // [NN, LLEN, DP] bf16
                              const unsigned short* __restrict__ Wih,   // [G4, DP] bf16
                              const unsigned short* __restrict__ Whh,   // [G4, EE] bf16
                              const float*          __restrict__ bias,  // [G4] (bih+bhh)
                              const unsigned short* __restrict__ h_prev,// [NN, EE] bf16
                              unsigned short*       __restrict__ h_next,// [NN, EE] bf16
                              float*                __restrict__ c_state,//[NN, EE] f32
                              unsigned short*       __restrict__ enc,   // [NN, LLEN, 2*EE] bf16
                              int t, int dir_off) {
  extern __shared__ unsigned short smem[];
  unsigned short* Bhh = smem;                       // [4*16][HH_STR]
  unsigned short* Bih = smem + 4 * 16 * HH_STR;     // [4*16][IH_STR]

  int tid = threadIdx.x;
  int jt = blockIdx.x;                 // 0..31

  // ---- cooperative staging of this block's weight slice (uniform, no guards)
  // Whh: 64 rows x 64 uint4  -> 4096 uint4 / 256 threads = 16 iters
  for (int f = tid; f < 64 * 64; f += 256) {
    int row = f >> 6;                  // 0..63  (= g*16 + r)
    int q   = f & 63;                  // uint4 index within row
    int g = row >> 4, r = row & 15;
    const uint4* src = reinterpret_cast<const uint4*>(
        Whh + (size_t)(g * EE + jt * 16 + r) * EE) + q;
    *(reinterpret_cast<uint4*>(Bhh + row * HH_STR) + q) = *src;
  }
  // Wih: 64 rows x 40 uint4 -> 2560 uint4 / 256 threads = 10 iters
  for (int f = tid; f < 64 * 40; f += 256) {
    int row = f / 40;
    int q   = f % 40;
    int g = row >> 4, r = row & 15;
    const uint4* src = reinterpret_cast<const uint4*>(
        Wih + (size_t)(g * EE + jt * 16 + r) * DP) + q;
    *(reinterpret_cast<uint4*>(Bih + row * IH_STR) + q) = *src;
  }
  __syncthreads();

  int wave = tid >> 5;                 // 0..7
  int lane = tid & 31;
  int mt = blockIdx.y * 8 + wave;      // 0..15
  int lm = lane & 15;
  int hi = lane >> 4;                  // 0 or 1
  int row_m = mt * 16 + lm;            // A-matrix row for this lane
  int n_j   = jt * 16 + lm;            // gate-local output column for this lane

  v8f acc[4] = {};

  // ---- recurrent part: K = EE = 512 (16 chunks of 32) ----
  const unsigned short* arow_h = h_prev + (size_t)row_m * EE;
  for (int kc = 0; kc < EE; kc += 32) {
    int bk = kc + hi * 8;
    v16bf a = load_frag(arow_h + bk);
#pragma unroll
    for (int g = 0; g < 4; ++g) {
      v16bf b = load_frag(Bhh + (g * 16 + lm) * HH_STR + bk);
      acc[g] = __builtin_amdgcn_wmma_f32_16x16x32_bf16(false, a, false, b,
                                                       (short)0, acc[g], false, false);
    }
  }

  // ---- input part: K = DP = 320 (10 chunks of 32) ----
  const unsigned short* arow_x = emb + ((size_t)row_m * LLEN + t) * DP;
  for (int kc = 0; kc < DP; kc += 32) {
    int bk = kc + hi * 8;
    v16bf a = load_frag(arow_x + bk);
#pragma unroll
    for (int g = 0; g < 4; ++g) {
      v16bf b = load_frag(Bih + (g * 16 + lm) * IH_STR + bk);
      acc[g] = __builtin_amdgcn_wmma_f32_16x16x32_bf16(false, a, false, b,
                                                       (short)0, acc[g], false, false);
    }
  }

  // ---- gate nonlinearities + state update (torch order i, f, g, o) ----
  float bi = bias[0 * EE + n_j];
  float bf = bias[1 * EE + n_j];
  float bg = bias[2 * EE + n_j];
  float bo = bias[3 * EE + n_j];
#pragma unroll
  for (int r = 0; r < 8; ++r) {
    int m = mt * 16 + hi * 8 + r;            // C/D layout: VGPR r -> M = r + 8*hi
    float gi = acc[0][r] + bi;
    float gf = acc[1][r] + bf;
    float gg = acc[2][r] + bg;
    float go = acc[3][r] + bo;
    size_t cidx = (size_t)m * EE + n_j;
    float c_old = c_state[cidx];
    float cn = sigm(gf) * c_old + sigm(gi) * tanhf(gg);
    float hn = sigm(go) * tanhf(cn);
    c_state[cidx] = cn;
    unsigned short hb = f2bf(hn);
    h_next[cidx] = hb;
    enc[((size_t)m * LLEN + t) * (2 * EE) + dir_off + n_j] = hb;
  }
}

// ---------------------------------------------------------------------------
// Generic WMMA GEMM: C[M,N] = A[M,K] * B[N,K]^T + bias[N], bf16 in, f32 out.
// One wave per 16x16 output tile. K must be a multiple of 32, M/N of 16.
// ---------------------------------------------------------------------------
__global__ void gemm_bf16_bias(const unsigned short* __restrict__ A,
                               const unsigned short* __restrict__ B,
                               const float*          __restrict__ bias,
                               float*                __restrict__ C,
                               int Mt, int Nt, int K) {
  int wave = (blockIdx.x * blockDim.x + threadIdx.x) >> 5;
  if (wave >= Mt * Nt) return;               // wave-uniform, EXEC stays all-ones
  int lane = threadIdx.x & 31;
  int mt = wave / Nt;
  int nt = wave % Nt;
  int lm = lane & 15;
  int hi = lane >> 4;
  const unsigned short* arow = A + (size_t)(mt * 16 + lm) * K;
  const unsigned short* brow = B + (size_t)(nt * 16 + lm) * K;
  v8f acc = {};
  for (int kc = 0; kc < K; kc += 32) {
    int bk = kc + hi * 8;
    v16bf a = load_frag(arow + bk);
    v16bf b = load_frag(brow + bk);
    acc = __builtin_amdgcn_wmma_f32_16x16x32_bf16(false, a, false, b,
                                                  (short)0, acc, false, false);
  }
  int n = nt * 16 + lm;
  float bv = bias[n];
  int Ncols = Nt * 16;
#pragma unroll
  for (int r = 0; r < 8; ++r) {
    int m = mt * 16 + hi * 8 + r;
    C[(size_t)m * Ncols + n] = acc[r] + bv;
  }
}

// ---------------------------------------------------------------------------
// Attention collapse: CW[h,e] = sum_a ctx[h,a]*W[a,e];  cb[h] = ctx[h,:]@b
// (no nonlinearity between the two attention projections in the reference)
// ---------------------------------------------------------------------------
__global__ void build_cw(const float* __restrict__ W,    // [AHID, 2*EE]
                         const float* __restrict__ b,    // [AHID]
                         const float* __restrict__ ctx,  // [HOPS, AHID]
                         float* __restrict__ CW,         // [HOPS, 2*EE]
                         float* __restrict__ cb) {       // [HOPS]
  int tid = threadIdx.x;
  for (int idx = tid; idx < HOPS * 2 * EE; idx += blockDim.x) {
    int h = idx / (2 * EE);
    int e = idx % (2 * EE);
    float s = 0.0f;
    for (int a = 0; a < AHID; ++a) s += ctx[h * AHID + a] * W[a * (2 * EE) + e];
    CW[idx] = s;
  }
  if (tid < HOPS) {
    float s = 0.0f;
    for (int a = 0; a < AHID; ++a) s += ctx[tid * AHID + a] * b[a];
    cb[tid] = s;
  }
}

// scores = enc @ CW^T + cb; softmax over HOPS; multiply by length mask.
__global__ void attn_scores(const unsigned short* __restrict__ enc,  // [NN, LLEN, 2*EE] bf16
                            const float* __restrict__ CW, const float* __restrict__ cb,
                            const int* __restrict__ lengths,
                            float* __restrict__ attnm) {             // [NN, LLEN, HOPS]
  int id = blockIdx.x * blockDim.x + threadIdx.x;  // n*LLEN + l
  if (id >= NN * LLEN) return;
  int n = id / LLEN;
  int l = id % LLEN;
  float s0 = cb[0], s1 = cb[1], s2 = cb[2], s3 = cb[3];
  const unsigned short* row = enc + (size_t)id * (2 * EE);
  for (int e = 0; e < 2 * EE; ++e) {
    float v = bf2f(row[e]);
    s0 += v * CW[e];
    s1 += v * CW[2 * EE + e];
    s2 += v * CW[2 * 2 * EE + e];
    s3 += v * CW[3 * 2 * EE + e];
  }
  float mx = fmaxf(fmaxf(s0, s1), fmaxf(s2, s3));
  float e0 = __expf(s0 - mx), e1 = __expf(s1 - mx), e2 = __expf(s2 - mx), e3 = __expf(s3 - mx);
  float inv = 1.0f / (e0 + e1 + e2 + e3);
  float mask = (l < lengths[n]) ? 1.0f : 0.0f;
  float* out = attnm + (size_t)id * HOPS;
  out[0] = e0 * inv * mask;
  out[1] = e1 * inv * mask;
  out[2] = e2 * inv * mask;
  out[3] = e3 * inv * mask;
}

// sent[n, e*HOPS + h] = sum_l enc[n,l,e] * attnm[n,l,h]   (bf16 output for GEMM)
__global__ void attn_pool(const unsigned short* __restrict__ enc,
                          const float* __restrict__ attnm,
                          unsigned short* __restrict__ sent_bf) {    // [NN, FF]
  int id = blockIdx.x * blockDim.x + threadIdx.x;  // n*(2*EE) + e
  if (id >= NN * 2 * EE) return;
  int n = id / (2 * EE);
  int e = id % (2 * EE);
  float a0 = 0.f, a1 = 0.f, a2 = 0.f, a3 = 0.f;
  for (int l = 0; l < LLEN; ++l) {
    float v = bf2f(enc[((size_t)n * LLEN + l) * (2 * EE) + e]);
    const float* w = attnm + ((size_t)n * LLEN + l) * HOPS;
    a0 += v * w[0]; a1 += v * w[1]; a2 += v * w[2]; a3 += v * w[3];
  }
  size_t base = (size_t)n * FF + (size_t)e * HOPS;
  sent_bf[base + 0] = f2bf(a0);
  sent_bf[base + 1] = f2bf(a1);
  sent_bf[base + 2] = f2bf(a2);
  sent_bf[base + 3] = f2bf(a3);
}

// ---------------------------------------------------------------------------
// Sentence BiLSTM recurrence: batch 1, 256 sequential steps. Latency-bound ->
// single kernel, one block per direction, h in LDS, c in registers.
// xg_* already contain x@Wih^T + bih + bhh (from gemm_bf16_bias).
// ---------------------------------------------------------------------------
__global__ void sent_lstm(const float* __restrict__ xg_f,   // [NN, G4]
                          const float* __restrict__ xg_b,   // [NN, G4]
                          const float* __restrict__ Whh_f,  // [G4, EE] f32
                          const float* __restrict__ Whh_b,  // [G4, EE] f32
                          float* __restrict__ enc2) {       // [NN, 2*EE]
  __shared__ float h_lds[EE];
  __shared__ float g_lds[G4];
  int dir = blockIdx.x;
  const float* xg  = dir ? xg_b  : xg_f;
  const float* Whh = dir ? Whh_b : Whh_f;
  int tid = threadIdx.x;
  if (tid < EE) h_lds[tid] = 0.0f;
  float c_reg = 0.0f;
  __syncthreads();
  for (int s = 0; s < NN; ++s) {
    int t = dir ? (NN - 1 - s) : s;
    for (int n = tid; n < G4; n += blockDim.x) {
      float acc = xg[(size_t)t * G4 + n];
      const float* wr = Whh + (size_t)n * EE;
      for (int k = 0; k < EE; ++k) acc += h_lds[k] * wr[k];
      g_lds[n] = acc;
    }
    __syncthreads();
    if (tid < EE) {
      float gi = g_lds[tid];
      float gf = g_lds[EE + tid];
      float gg = g_lds[2 * EE + tid];
      float go = g_lds[3 * EE + tid];
      c_reg = sigm(gf) * c_reg + sigm(gi) * tanhf(gg);
      float hn = sigm(go) * tanhf(c_reg);
      h_lds[tid] = hn;
      enc2[(size_t)t * (2 * EE) + dir * EE + tid] = hn;
    }
    __syncthreads();
  }
}

// out[n, c] = enc2[n, :] @ lin_W[c, :] + lin_b[c]
__global__ void final_linear(const float* __restrict__ enc2,
                             const float* __restrict__ W,   // [NLAB, 2*EE]
                             const float* __restrict__ b,
                             float* __restrict__ out) {     // [NN, NLAB]
  int id = blockIdx.x * blockDim.x + threadIdx.x;
  if (id >= NN * NLAB) return;
  int n = id / NLAB;
  int c = id % NLAB;
  float s = b[c];
  for (int e = 0; e < 2 * EE; ++e) s += enc2[(size_t)n * (2 * EE) + e] * W[(size_t)c * (2 * EE) + e];
  out[id] = s;
}

// ---------------------------------------------------------------------------
// host launcher
// ---------------------------------------------------------------------------
static inline size_t align256(size_t x) { return (x + 255) & ~(size_t)255; }

extern "C" void kernel_launch(void* const* d_in, const int* in_sizes, int n_in,
                              void* d_out, int out_size, void* d_ws, size_t ws_size,
                              hipStream_t stream) {
  (void)in_sizes; (void)n_in; (void)out_size; (void)ws_size;

  const int*   tokens    = (const int*)  d_in[0];
  const int*   lengths   = (const int*)  d_in[1];
  const float* embedding = (const float*)d_in[2];
  const float* tokWihF   = (const float*)d_in[3];
  const float* tokWhhF   = (const float*)d_in[4];
  const float* tokBihF   = (const float*)d_in[5];
  const float* tokBhhF   = (const float*)d_in[6];
  const float* tokWihB   = (const float*)d_in[7];
  const float* tokWhhB   = (const float*)d_in[8];
  const float* tokBihB   = (const float*)d_in[9];
  const float* tokBhhB   = (const float*)d_in[10];
  const float* attnW     = (const float*)d_in[11];
  const float* attnB     = (const float*)d_in[12];
  const float* attnCtx   = (const float*)d_in[13];
  const float* sentWihF  = (const float*)d_in[14];
  const float* sentWhhF  = (const float*)d_in[15];
  const float* sentBihF  = (const float*)d_in[16];
  const float* sentBhhF  = (const float*)d_in[17];
  const float* sentWihB  = (const float*)d_in[18];
  const float* sentWhhB  = (const float*)d_in[19];
  const float* sentBihB  = (const float*)d_in[20];
  const float* sentBhhB  = (const float*)d_in[21];
  const float* linW      = (const float*)d_in[22];
  const float* linB      = (const float*)d_in[23];
  float* out = (float*)d_out;

  // ---- workspace carve-up (~138 MB) ----
  char* ws = (char*)d_ws;
  size_t off = 0;
  auto take = [&](size_t bytes) { char* p = ws + off; off = align256(off + bytes); return p; };

  unsigned short* emb_bf   = (unsigned short*)take((size_t)NN * LLEN * DP * 2);  // 21 MB
  unsigned short* WihF_bf  = (unsigned short*)take((size_t)G4 * DP * 2);
  unsigned short* WihB_bf  = (unsigned short*)take((size_t)G4 * DP * 2);
  unsigned short* WhhF_bf  = (unsigned short*)take((size_t)G4 * EE * 2);
  unsigned short* WhhB_bf  = (unsigned short*)take((size_t)G4 * EE * 2);
  float*          biasF    = (float*)take((size_t)G4 * 4);
  float*          biasB    = (float*)take((size_t)G4 * 4);
  unsigned short* hF[2]    = {(unsigned short*)take((size_t)NN * EE * 2),
                              (unsigned short*)take((size_t)NN * EE * 2)};
  unsigned short* hB[2]    = {(unsigned short*)take((size_t)NN * EE * 2),
                              (unsigned short*)take((size_t)NN * EE * 2)};
  float*          cF       = (float*)take((size_t)NN * EE * 4);
  float*          cB       = (float*)take((size_t)NN * EE * 4);
  unsigned short* enc_bf   = (unsigned short*)take((size_t)NN * LLEN * 2 * EE * 2); // 67 MB
  float*          CW       = (float*)take((size_t)HOPS * 2 * EE * 4);
  float*          cb       = (float*)take((size_t)HOPS * 4);
  float*          attnm    = (float*)take((size_t)NN * LLEN * HOPS * 4);
  unsigned short* sent_bf  = (unsigned short*)take((size_t)NN * FF * 2);
  unsigned short* sWihF_bf = (unsigned short*)take((size_t)G4 * FF * 2);            // 16 MB
  unsigned short* sWihB_bf = (unsigned short*)take((size_t)G4 * FF * 2);            // 16 MB
  float*          sbiasF   = (float*)take((size_t)G4 * 4);
  float*          sbiasB   = (float*)take((size_t)G4 * 4);
  float*          xg_sf    = (float*)take((size_t)NN * G4 * 4);
  float*          xg_sb    = (float*)take((size_t)NN * G4 * 4);
  float*          enc2     = (float*)take((size_t)NN * 2 * EE * 4);

  const int T = 256;
  auto blocks = [](long n, int t) { return (int)((n + t - 1) / t); };

  // ---- stage 0: conversions / init ----
  embed_pad<<<blocks((long)NN * LLEN * DP, T), T, 0, stream>>>(tokens, embedding, emb_bf);
  cvt_pad_bf16<<<blocks((long)G4 * DP, T), T, 0, stream>>>(tokWihF, WihF_bf, G4);
  cvt_pad_bf16<<<blocks((long)G4 * DP, T), T, 0, stream>>>(tokWihB, WihB_bf, G4);
  cvt_bf16<<<blocks((long)G4 * EE, T), T, 0, stream>>>(tokWhhF, WhhF_bf, (long)G4 * EE);
  cvt_bf16<<<blocks((long)G4 * EE, T), T, 0, stream>>>(tokWhhB, WhhB_bf, (long)G4 * EE);
  add_vec<<<blocks(G4, T), T, 0, stream>>>(tokBihF, tokBhhF, biasF, G4);
  add_vec<<<blocks(G4, T), T, 0, stream>>>(tokBihB, tokBhhB, biasB, G4);
  fill_zero_u32<<<blocks((long)NN * EE / 2, T), T, 0, stream>>>((unsigned int*)hF[0], (long)NN * EE / 2);
  fill_zero_u32<<<blocks((long)NN * EE / 2, T), T, 0, stream>>>((unsigned int*)hB[0], (long)NN * EE / 2);
  fill_zero_u32<<<blocks((long)NN * EE, T), T, 0, stream>>>((unsigned int*)cF, (long)NN * EE);
  fill_zero_u32<<<blocks((long)NN * EE, T), T, 0, stream>>>((unsigned int*)cB, (long)NN * EE);

  // ---- stage 1: token BiLSTM, 128 sequential fused WMMA steps per direction ----
  dim3 step_grid(32, 2);   // jt x m-groups; 256 threads = 8 waves per block
  for (int s = 0; s < LLEN; ++s) {
    tok_lstm_step<<<step_grid, 256, LDS_BYTES, stream>>>(emb_bf, WihF_bf, WhhF_bf, biasF,
                                                         hF[s & 1], hF[(s + 1) & 1], cF, enc_bf,
                                                         /*t=*/s, /*dir_off=*/0);
    tok_lstm_step<<<step_grid, 256, LDS_BYTES, stream>>>(emb_bf, WihB_bf, WhhB_bf, biasB,
                                                         hB[s & 1], hB[(s + 1) & 1], cB, enc_bf,
                                                         /*t=*/LLEN - 1 - s, /*dir_off=*/EE);
  }

  // ---- stage 2: attention (collapsed to one thin GEMV + softmax + pool) ----
  build_cw<<<1, 256, 0, stream>>>(attnW, attnB, attnCtx, CW, cb);
  attn_scores<<<blocks((long)NN * LLEN, T), T, 0, stream>>>(enc_bf, CW, cb, lengths, attnm);
  attn_pool<<<blocks((long)NN * 2 * EE, T), T, 0, stream>>>(enc_bf, attnm, sent_bf);

  // ---- stage 3: sentence BiLSTM ----
  cvt_bf16<<<blocks((long)G4 * FF, T), T, 0, stream>>>(sentWihF, sWihF_bf, (long)G4 * FF);
  cvt_bf16<<<blocks((long)G4 * FF, T), T, 0, stream>>>(sentWihB, sWihB_bf, (long)G4 * FF);
  add_vec<<<blocks(G4, T), T, 0, stream>>>(sentBihF, sentBhhF, sbiasF, G4);
  add_vec<<<blocks(G4, T), T, 0, stream>>>(sentBihB, sentBhhB, sbiasB, G4);
  // xg = sent @ Wih^T + bias : [256,4096] x [2048,4096]^T -> [256,2048]
  {
    int Mt = NN / 16, Nt = G4 / 16;                // 16 x 128 = 2048 waves
    int nblk = (Mt * Nt + 3) / 4;                  // 4 waves / 128-thread block
    gemm_bf16_bias<<<nblk, 128, 0, stream>>>(sent_bf, sWihF_bf, sbiasF, xg_sf, Mt, Nt, FF);
    gemm_bf16_bias<<<nblk, 128, 0, stream>>>(sent_bf, sWihB_bf, sbiasB, xg_sb, Mt, Nt, FF);
  }
  sent_lstm<<<2, 1024, 0, stream>>>(xg_sf, xg_sb, sentWhhF, sentWhhB, enc2);

  // ---- stage 4: classifier head ----
  final_linear<<<blocks((long)NN * NLAB, T), T, 0, stream>>>(enc2, linW, linB, out);
}
